// VGG11_18150531793023
// MI455X (gfx1250) — 8-bit matrix operand order repaired and checked by reading, compiled, not run on hardware
//
#include <hip/hip_runtime.h>
#include <hip/hip_bf16.h>
#include <stddef.h>


typedef int v8i __attribute__((ext_vector_type(8)));

__device__ __forceinline__ signed char sgn8(float v) {
    return v > 0.f ? (signed char)1 : (v < 0.f ? (signed char)-1 : (signed char)0);
}

__device__ __forceinline__ v8i wmma_iu8(v8i a, v8i b, v8i c) {
    return __builtin_amdgcn_wmma_i32_16x16x64_iu8(true, a, true, b, c, false, false);
}

__device__ __forceinline__ v8i loadA_frag(const signed char* pp, bool valid) {
    v8i a = {0, 0, 0, 0, 0, 0, 0, 0};
    if (valid) {
        int2 q0 = *(const int2*)(pp);
        int2 q1 = *(const int2*)(pp + 16);
        int2 q2 = *(const int2*)(pp + 32);
        int2 q3 = *(const int2*)(pp + 48);
        a[0] = q0.x; a[1] = q0.y; a[2] = q1.x; a[3] = q1.y;
        a[4] = q2.x; a[5] = q2.y; a[6] = q3.x; a[7] = q3.y;
    }
    return a;
}

__device__ __forceinline__ v8i loadB_frag(const signed char* bp) {
    int4 r0 = *(const int4*)(bp);
    int4 r1 = *(const int4*)(bp + 16);
    v8i b;
    b[0] = r0.x; b[1] = r0.y; b[2] = r0.z; b[3] = r0.w;
    b[4] = r1.x; b[5] = r1.y; b[6] = r1.z; b[7] = r1.w;
    return b;
}

template <int CIN, int COUT>
__global__ void packw_kernel(const float* __restrict__ w, signed char* __restrict__ out) {
    constexpr int KCH = CIN / 64, NT = COUT / 16;
    int tid = blockIdx.x * blockDim.x + threadIdx.x;
    constexpr int total = 9 * KCH * NT * 1024;
    if (tid >= total) return;
    int byt  = tid & 3;
    int v    = (tid >> 2) & 7;
    int lane = (tid >> 5) & 31;
    int blk  = tid >> 10;
    int nc   = blk % NT;
    int kc   = (blk / NT) % KCH;
    int tap  = blk / (NT * KCH);
    int hi = lane >> 4, li = lane & 15;
    int K = (v >> 1) * 16 + hi * 8 + (v & 1) * 4 + byt;
    int i = kc * 64 + K;
    int o = nc * 16 + li;
    float wv = w[(size_t)(o * CIN + i) * 9 + tap];
    out[tid] = sgn8(wv);
}

__global__ void conv1_kernel(const float* __restrict__ x, const float* __restrict__ w1,
                             const float* __restrict__ b1, signed char* __restrict__ out) {
    __shared__ float patch[48];
    int p  = blockIdx.x;
    int x2 = p % 112;
    int y2 = (p / 112) % 112;
    int n  = p / (112 * 112);
    int tid = threadIdx.x;
    if (tid < 48) {
        int ci = tid / 16, r = (tid / 4) & 3, cc = tid & 3;
        int iy = y2 * 2 - 1 + r, ix = x2 * 2 - 1 + cc;
        float v = 0.f;
        if (iy >= 0 && iy < 224 && ix >= 0 && ix < 224)
            v = x[((size_t)(n * 3 + ci) * 224 + iy) * 224 + ix];
        patch[tid] = v;
    }
    __syncthreads();
    int c = tid;
    float ws[27];
#pragma unroll
    for (int t = 0; t < 27; ++t) {
        float w = w1[c * 27 + t];
        ws[t] = w > 0.f ? 1.f : (w < 0.f ? -1.f : 0.f);
    }
    float mx = -3.0e38f;
#pragma unroll
    for (int py = 0; py < 2; ++py)
#pragma unroll
        for (int qx = 0; qx < 2; ++qx) {
            float s = 0.f;
#pragma unroll
            for (int ci = 0; ci < 3; ++ci)
#pragma unroll
                for (int ky = 0; ky < 3; ++ky)
#pragma unroll
                    for (int kx = 0; kx < 3; ++kx)
                        s += patch[ci * 16 + (py + ky) * 4 + (qx + kx)] * ws[(ci * 3 + ky) * 3 + kx];
            mx = fmaxf(mx, s);
        }
    out[(size_t)p * 64 + c] = sgn8(mx + b1[c]);
}

template <int COUT, int H, int W, bool F32OUT>
__device__ __forceinline__ void store_tile(v8i acc, int mt, int nc, int hi, int li,
                                           const float* __restrict__ bias,
                                           void* __restrict__ Out) {
#pragma unroll
    for (int r = 0; r < 8; ++r) {
        int mm = mt * 16 + r + hi * 8;
        int ox = mm % W, oy = (mm / W) % H, on = mm / (W * H);
        int c  = nc * 16 + li;
        size_t idx = (size_t)((on * H + oy) * W + ox) * COUT + c;
        float v = (float)acc[r] + bias[c];
        if (F32OUT)
            ((float*)Out)[idx] = v;
        else
            ((signed char*)Out)[idx] = sgn8(v);
    }
}

template <int CIN, int COUT, int H, int W, bool F32OUT>
__global__ void binconv3x3_kernel(const signed char* __restrict__ A,
                                  const signed char* __restrict__ Bp,
                                  const float* __restrict__ bias,
                                  void* __restrict__ Out, int N) {
    constexpr int KCH = CIN / 64;
    constexpr int NT  = COUT / 16;
    constexpr int NP  = NT / 2;
    const int npix      = N * H * W;
    const int mTiles    = npix / 16;
    const int totalJobs = (mTiles / 2) * NP;
    int job = blockIdx.x * (blockDim.x >> 5) + (threadIdx.x >> 5);
    if (job >= totalJobs) return;
    int lane = threadIdx.x & 31;
    int jn = job % NP;
    int jm = job / NP;
    int hi = lane >> 4, li = lane & 15;
    int nc0 = jn * 2, mt0 = jm * 2;

    int m0 = mt0 * 16 + li;
    int m1 = m0 + 16;
    int px0 = m0 % W, py0 = (m0 / W) % H, pn0 = m0 / (W * H);
    int px1 = m1 % W, py1 = (m1 / W) % H, pn1 = m1 / (W * H);

    v8i acc00 = {0, 0, 0, 0, 0, 0, 0, 0};
    v8i acc01 = {0, 0, 0, 0, 0, 0, 0, 0};
    v8i acc10 = {0, 0, 0, 0, 0, 0, 0, 0};
    v8i acc11 = {0, 0, 0, 0, 0, 0, 0, 0};

    for (int tap = 0; tap < 9; ++tap) {
        int dy = tap / 3 - 1, dx = tap % 3 - 1;
        int iy0 = py0 + dy, ix0 = px0 + dx;
        int iy1 = py1 + dy, ix1 = px1 + dx;
        bool v0 = (iy0 >= 0) && (iy0 < H) && (ix0 >= 0) && (ix0 < W);
        bool v1 = (iy1 >= 0) && (iy1 < H) && (ix1 >= 0) && (ix1 < W);
        const signed char* ap0 = A + (size_t)((pn0 * H + iy0) * W + ix0) * CIN + hi * 8;
        const signed char* ap1 = A + (size_t)((pn1 * H + iy1) * W + ix1) * CIN + hi * 8;
        const signed char* bbase = Bp + (size_t)(tap * KCH) * NT * 1024;
#pragma unroll
        for (int kc = 0; kc < KCH; ++kc) {
            v8i a0 = loadA_frag(ap0 + kc * 64, v0);
            v8i a1 = loadA_frag(ap1 + kc * 64, v1);
            const signed char* bp0 = bbase + (size_t)(kc * NT + nc0) * 1024 + lane * 32;
            v8i b0 = loadB_frag(bp0);
            v8i b1 = loadB_frag(bp0 + 1024);
            if (tap != 8) {
                __builtin_prefetch(bp0 + (size_t)NT * 1024, 0, 1);
                __builtin_prefetch(bp0 + (size_t)NT * 1024 + 1024, 0, 1);
            }
            acc00 = wmma_iu8(a0, b0, acc00);
            acc01 = wmma_iu8(a0, b1, acc01);
            acc10 = wmma_iu8(a1, b0, acc10);
            acc11 = wmma_iu8(a1, b1, acc11);
        }
    }
    store_tile<COUT, H, W, F32OUT>(acc00, mt0,     nc0,     hi, li, bias, Out);
    store_tile<COUT, H, W, F32OUT>(acc01, mt0,     nc0 + 1, hi, li, bias, Out);
    store_tile<COUT, H, W, F32OUT>(acc10, mt0 + 1, nc0,     hi, li, bias, Out);
    store_tile<COUT, H, W, F32OUT>(acc11, mt0 + 1, nc0 + 1, hi, li, bias, Out);
}

__global__ void maxpool8_kernel(const signed char* __restrict__ in, signed char* __restrict__ out,
                                int N, int H, int W, int C) {
    size_t tid = (size_t)blockIdx.x * blockDim.x + threadIdx.x;
    size_t total = (size_t)N * H * W * C;
    if (tid >= total) return;
    int c = (int)(tid % C);
    size_t pp = tid / C;
    int x = (int)(pp % W);
    int y = (int)((pp / W) % H);
    int n = (int)(pp / ((size_t)W * H));
    size_t base = ((size_t)(n * (2 * H) + 2 * y) * (2 * W) + 2 * x) * C + c;
    signed char a0 = in[base], a1 = in[base + C];
    signed char a2 = in[base + (size_t)2 * W * C], a3 = in[base + (size_t)2 * W * C + C];
    signed char m0 = a0 > a1 ? a0 : a1;
    signed char m1 = a2 > a3 ? a2 : a3;
    out[tid] = m0 > m1 ? m0 : m1;
}

__global__ void maxpoolf_kernel(const float* __restrict__ in, float* __restrict__ out,
                                int N, int H, int W, int C) {
    size_t tid = (size_t)blockIdx.x * blockDim.x + threadIdx.x;
    size_t total = (size_t)N * H * W * C;
    if (tid >= total) return;
    int c = (int)(tid % C);
    size_t pp = tid / C;
    int x = (int)(pp % W);
    int y = (int)((pp / W) % H);
    int n = (int)(pp / ((size_t)W * H));
    size_t base = ((size_t)(n * (2 * H) + 2 * y) * (2 * W) + 2 * x) * C + c;
    float a0 = in[base], a1 = in[base + C];
    float a2 = in[base + (size_t)2 * W * C], a3 = in[base + (size_t)2 * W * C + C];
    out[tid] = fmaxf(fmaxf(a0, a1), fmaxf(a2, a3));
}

__global__ void avgsign_kernel(const float* __restrict__ in, signed char* __restrict__ out) {
    int tid = blockIdx.x * blockDim.x + threadIdx.x;
    const int total = 32 * 7 * 7 * 512;
    if (tid >= total) return;
    int c  = tid % 512;
    int xx = (tid / 512) % 7;
    int yy = (tid / (512 * 7)) % 7;
    int n  = tid / (512 * 49);
    size_t base = ((size_t)(n * 14 + 2 * yy) * 14 + 2 * xx) * 512 + c;
    float s = in[base] + in[base + 512] + in[base + 14 * 512] + in[base + 14 * 512 + 512];
    out[(size_t)n * 25088 + c * 49 + yy * 7 + xx] = sgn8(s);
}

__global__ void fc1_kernel(const signed char* __restrict__ A, const float* __restrict__ Wf,
                           float* __restrict__ Part) {
    const int K = 25088, NOUT = 4096;
    const int nTiles = NOUT / 16, mTiles = 2, KS = 8, KCPS = (K / 64) / KS;
    int job = blockIdx.x * (blockDim.x >> 5) + (threadIdx.x >> 5);
    if (job >= nTiles * mTiles * KS) return;
    int lane = threadIdx.x & 31;
    int ks   = job % KS;
    int rest = job / KS;
    int nc = rest % nTiles, mt = rest / nTiles;
    int hi = lane >> 4, li = lane & 15;
    const signed char* ap0 = A + (size_t)(mt * 16 + li) * K + hi * 8;
    const float* wrow = Wf + (size_t)(nc * 16 + li) * K;
    v8i acc = {0, 0, 0, 0, 0, 0, 0, 0};
    for (int kk = 0; kk < KCPS; ++kk) {
        int kc = ks * KCPS + kk;
        v8i a = loadA_frag(ap0 + kc * 64, true);
        v8i b;
#pragma unroll
        for (int g = 0; g < 2; ++g) {
            const float* wp = wrow + kc * 64 + g * 32 + hi * 8;
#pragma unroll
            for (int vv = 0; vv < 4; ++vv) {
                float4 f = *(const float4*)(wp + (vv >> 1) * 16 + (vv & 1) * 4);
                unsigned d;
                d  = (unsigned)(unsigned char)sgn8(f.x);
                d |= (unsigned)(unsigned char)sgn8(f.y) << 8;
                d |= (unsigned)(unsigned char)sgn8(f.z) << 16;
                d |= (unsigned)(unsigned char)sgn8(f.w) << 24;
                b[g * 4 + vv] = (int)d;
            }
        }
        acc = wmma_iu8(a, b, acc);
    }
#pragma unroll
    for (int r = 0; r < 8; ++r) {
        int bb = mt * 16 + r + hi * 8;
        int jj = nc * 16 + li;
        Part[((size_t)ks * 32 + bb) * NOUT + jj] = (float)acc[r];
    }
}

__global__ void fc1_reduce_kernel(const float* __restrict__ Part, const float* __restrict__ bias,
                                  float* __restrict__ Out) {
    int tid = blockIdx.x * blockDim.x + threadIdx.x;
    if (tid >= 32 * 4096) return;
    int j = tid % 4096;
    float s = bias[j];
#pragma unroll
    for (int ks = 0; ks < 8; ++ks) s += Part[(size_t)ks * 32 * 4096 + tid];
    Out[tid] = s;
}

__global__ void fc2_kernel(const float* __restrict__ h, const float* __restrict__ w,
                           const float* __restrict__ b, float* __restrict__ out) {
    int tid = blockIdx.x * blockDim.x + threadIdx.x;
    if (tid >= 320) return;
    int bb = tid / 10, j = tid % 10;
    const float* hp = h + bb * 4096;
    const float* wp = w + j * 4096;
    float s = 0.f;
    for (int k = 0; k < 4096; ++k) s += hp[k] * wp[k];
    out[tid] = s + b[j];
}

extern "C" void kernel_launch(void* const* d_in, const int* in_sizes, int n_in,
                              void* d_out, int out_size, void* d_ws, size_t ws_size,
                              hipStream_t stream) {
    const float* x    = (const float*)d_in[0];
    const float* w1   = (const float*)d_in[1];
    const float* b1   = (const float*)d_in[2];
    const float* w2   = (const float*)d_in[3];
    const float* b2   = (const float*)d_in[4];
    const float* w3   = (const float*)d_in[5];
    const float* b3   = (const float*)d_in[6];
    const float* w4   = (const float*)d_in[7];
    const float* b4   = (const float*)d_in[8];
    const float* w5   = (const float*)d_in[9];
    const float* b5   = (const float*)d_in[10];
    const float* w6   = (const float*)d_in[11];
    const float* b6   = (const float*)d_in[12];
    const float* fc1w = (const float*)d_in[13];
    const float* fc1b = (const float*)d_in[14];
    const float* fc2w = (const float*)d_in[15];
    const float* fc2b = (const float*)d_in[16];
    float* out = (float*)d_out;
    char* ws = (char*)d_ws;

    const size_t WP2 = 0;
    const size_t WP3 = WP2 + 73728;
    const size_t WP4 = WP3 + 294912;
    const size_t WP5 = WP4 + 589824;
    const size_t WP6 = WP5 + 1179648;
    const size_t SLAB0 = WP6 + 2359296;
    const size_t SLAB1 = SLAB0 + (52u << 20);
    signed char* wp2 = (signed char*)(ws + WP2);
    signed char* wp3 = (signed char*)(ws + WP3);
    signed char* wp4 = (signed char*)(ws + WP4);
    signed char* wp5 = (signed char*)(ws + WP5);
    signed char* wp6 = (signed char*)(ws + WP6);

    signed char* A2  = (signed char*)(ws + SLAB0);
    signed char* C2  = (signed char*)(ws + SLAB1);
    signed char* A3  = (signed char*)(ws + SLAB0);
    signed char* C3  = (signed char*)(ws + SLAB1);
    signed char* C4  = (signed char*)(ws + SLAB0);
    signed char* A5  = (signed char*)(ws + SLAB1);
    signed char* C5  = (signed char*)(ws + SLAB0);
    float*       C6f = (float*)(ws + SLAB1);
    float*       P6f = (float*)(ws + SLAB0);
    signed char* A8  = (signed char*)(ws + SLAB1);
    float*       PART = (float*)(ws + SLAB0 + (16u << 20));
    float*       F1   = (float*)(ws + SLAB0 + (24u << 20));

    packw_kernel<64, 128><<<(9 * 64 * 128 + 255) / 256, 256, 0, stream>>>(w2, wp2);
    packw_kernel<128, 256><<<(9 * 128 * 256 + 255) / 256, 256, 0, stream>>>(w3, wp3);
    packw_kernel<256, 256><<<(9 * 256 * 256 + 255) / 256, 256, 0, stream>>>(w4, wp4);
    packw_kernel<256, 512><<<(9 * 256 * 512 + 255) / 256, 256, 0, stream>>>(w5, wp5);
    packw_kernel<512, 512><<<(9 * 512 * 512 + 255) / 256, 256, 0, stream>>>(w6, wp6);

    conv1_kernel<<<32 * 112 * 112, 64, 0, stream>>>(x, w1, b1, A2);

    binconv3x3_kernel<64, 128, 112, 112, false>
        <<<((32 * 112 * 112 / 32) * (128 / 32)) / 8, 256, 0, stream>>>(A2, wp2, b2, C2, 32);
    {
        size_t tot = (size_t)32 * 56 * 56 * 128;
        maxpool8_kernel<<<(unsigned)((tot + 255) / 256), 256, 0, stream>>>(C2, A3, 32, 56, 56, 128);
    }
    binconv3x3_kernel<128, 256, 56, 56, false>
        <<<((32 * 56 * 56 / 32) * (256 / 32)) / 8, 256, 0, stream>>>(A3, wp3, b3, C3, 32);
    binconv3x3_kernel<256, 256, 56, 56, false>
        <<<((32 * 56 * 56 / 32) * (256 / 32)) / 8, 256, 0, stream>>>(C3, wp4, b4, C4, 32);
    {
        size_t tot = (size_t)32 * 28 * 28 * 256;
        maxpool8_kernel<<<(unsigned)((tot + 255) / 256), 256, 0, stream>>>(C4, A5, 32, 28, 28, 256);
    }
    binconv3x3_kernel<256, 512, 28, 28, false>
        <<<((32 * 28 * 28 / 32) * (512 / 32)) / 8, 256, 0, stream>>>(A5, wp5, b5, C5, 32);
    binconv3x3_kernel<512, 512, 28, 28, true>
        <<<((32 * 28 * 28 / 32) * (512 / 32)) / 8, 256, 0, stream>>>(C5, wp6, b6, (void*)C6f, 32);
    {
        size_t tot = (size_t)32 * 14 * 14 * 512;
        maxpoolf_kernel<<<(unsigned)((tot + 255) / 256), 256, 0, stream>>>(C6f, P6f, 32, 14, 14, 512);
    }
    avgsign_kernel<<<(32 * 7 * 7 * 512 + 255) / 256, 256, 0, stream>>>(P6f, A8);

    fc1_kernel<<<(2 * 256 * 8) / 8, 256, 0, stream>>>(A8, fc1w, PART);
    fc1_reduce_kernel<<<(32 * 4096 + 255) / 256, 256, 0, stream>>>(PART, fc1b, F1);
    fc2_kernel<<<2, 160, 0, stream>>>(F1, fc2w, fc2b, out);
}
